// Attention_6150393168392
// MI455X (gfx1250) — matrix operand order repaired and checked by reading, not compiled, not run on hardware
//
#include <hip/hip_runtime.h>
#include <math.h>

typedef __attribute__((ext_vector_type(16))) __bf16 v16bf;
typedef __attribute__((ext_vector_type(8)))  float  v8f;

#define BB 32
#define SS 4096
#define HH 256

__device__ __forceinline__ unsigned short f32_to_bf16(float f) {
    unsigned int u = __float_as_uint(f);
    unsigned int r = u + 0x7FFFu + ((u >> 16) & 1u);   // round-to-nearest-even
    return (unsigned short)(r >> 16);
}

// pack hi16(f1):hi16(f0) into one dword with a single v_perm_b32 (truncating cvt)
__device__ __forceinline__ unsigned int pack_bf16x2(float f0, float f1) {
    return __builtin_amdgcn_perm(__float_as_uint(f1), __float_as_uint(f0),
                                 0x07060302u);
}

__device__ __forceinline__ float fast_exp2(float x) {
#if __has_builtin(__builtin_amdgcn_exp2f)
    return __builtin_amdgcn_exp2f(x);
#else
    return exp2f(x);
#endif
}
__device__ __forceinline__ float fast_rcp(float x) {
#if __has_builtin(__builtin_amdgcn_rcpf)
    return __builtin_amdgcn_rcpf(x);
#else
    return 1.0f / x;
#endif
}

// ---------------------------------------------------------------------------
// Kernel 1: repack W_e = W[:, H:2H] into bf16 per WMMA B-fragment layout:
//   Wb[((t*8 + ks)*32 + lane)*16 + e]
//   n = t*16 + (lane&15);  k = ks*32 + (lane<16 ? 0 : 16) + e
//   value = B[k][n] = W_e[n][k] = W[n, 256+k]
// ---------------------------------------------------------------------------
__global__ __launch_bounds__(256) void repack_we_kernel(
    const float* __restrict__ W, unsigned short* __restrict__ Wb) {
    int idx = blockIdx.x * 256 + threadIdx.x;     // 0 .. 65535
    int e    = idx & 15;
    int lane = (idx >> 4) & 31;
    int ks   = (idx >> 9) & 7;
    int t    = idx >> 12;
    int n = t * 16 + (lane & 15);
    int k = ks * 32 + ((lane < 16) ? 0 : 16) + e;
    Wb[idx] = f32_to_bf16(W[n * (2 * HH) + HH + k]);
}

// ---------------------------------------------------------------------------
// Kernel 2: hb[b,h] = b_attn[h] + sum_k hidden[b,k] * W[h,k]
// ---------------------------------------------------------------------------
__global__ __launch_bounds__(256) void hidden_proj_kernel(
    const float* __restrict__ hidden, const float* __restrict__ W,
    const float* __restrict__ b_attn, float* __restrict__ hb) {
    int b = blockIdx.x, h = threadIdx.x;
    __shared__ float hs[HH];
    hs[h] = hidden[b * HH + h];
    __syncthreads();
    float acc = b_attn[h];
    const float* wrow = W + (size_t)h * (2 * HH);
#pragma unroll 8
    for (int k = 0; k < HH; ++k) acc += hs[k] * wrow[k];
    hb[b * HH + h] = acc;
}

// ---------------------------------------------------------------------------
// Kernel 3: G = enc @ W_e^T (bf16 WMMA, f32 acc) fused with
//   e[b,s] = sum_h v[h] * tanh(G[b,s,h] + hb[b,h])
// Grid: 1024 WGs; WG = 128 M-rows (one b), 16 waves (512 threads).
// Wb (128 KB) staged in LDS once per WG; B frags via ds_load_b128 pairs.
// Wave: msub = wave>>1 (16-row subtile, 0..7), nhalf = wave&1 (8 N-tiles).
// ---------------------------------------------------------------------------
__global__ __launch_bounds__(512) void attn_energy_kernel(
    const float* __restrict__ enc, const unsigned short* __restrict__ Wb,
    const float* __restrict__ hb, const float* __restrict__ v,
    float* __restrict__ e_out) {
    const int tile = blockIdx.x;
    const int m0 = tile * 128;           // global row = b*S + s
    const int b  = m0 >> 12;
    const int s0 = m0 & (SS - 1);

    __shared__ __attribute__((aligned(32))) unsigned short wtile[65536]; // 128 KB
    __shared__ float e_lds[128];

    // stage full bf16 W_e fragment buffer into LDS (coalesced 16B chunks)
    {
        const uint4* src = (const uint4*)Wb;
        uint4* dst = (uint4*)wtile;
#pragma unroll
        for (int i = 0; i < 16; ++i)
            dst[threadIdx.x + 512 * i] = src[threadIdx.x + 512 * i];
    }
    if (threadIdx.x < 128) e_lds[threadIdx.x] = 0.0f;
    __syncthreads();

    const int wave  = threadIdx.x >> 5;   // 0..15
    const int lane  = threadIdx.x & 31;
    const int msub  = wave >> 1;          // 0..7: 16-row M subtile
    const int nhalf = wave & 1;           // 0..1: 8 N-tiles of 16 cols

    const int r  = lane & 15;
    const long gr = (long)m0 + msub * 16 + r;
    const float* arow = enc + gr * HH;
    const int khalf = (lane < 16) ? 0 : 16;  // contiguous k order: lanes 16-31 take k 16..31 of each step

    v8f acc[8];
#pragma unroll
    for (int t = 0; t < 8; ++t)
#pragma unroll
        for (int j = 0; j < 8; ++j) acc[t][j] = 0.0f;

#pragma unroll
    for (int ks = 0; ks < 8; ++ks) {
        const int k0 = ks * 32 + khalf;
        // A frag: elems 0-15 = K k0..k0+15
        const float4 x0 = *(const float4*)(arow + k0);
        const float4 x1 = *(const float4*)(arow + k0 + 4);
        const float4 x2 = *(const float4*)(arow + k0 + 8);
        const float4 x3 = *(const float4*)(arow + k0 + 12);
        union { v16bf v; unsigned int p[8]; } af;
        af.p[0] = pack_bf16x2(x0.x, x0.y);
        af.p[1] = pack_bf16x2(x0.z, x0.w);
        af.p[2] = pack_bf16x2(x1.x, x1.y);
        af.p[3] = pack_bf16x2(x1.z, x1.w);
        af.p[4] = pack_bf16x2(x2.x, x2.y);
        af.p[5] = pack_bf16x2(x2.z, x2.w);
        af.p[6] = pack_bf16x2(x3.x, x3.y);
        af.p[7] = pack_bf16x2(x3.z, x3.w);

#pragma unroll
        for (int t = 0; t < 8; ++t) {
            const int nt = nhalf * 8 + t;
            const v16bf bv = *reinterpret_cast<const v16bf*>(
                &wtile[(((nt * 8 + ks) * 32) + lane) * 16]);
            acc[t] = __builtin_amdgcn_wmma_f32_16x16x32_bf16(
                false, af.v, false, bv, (short)0, acc[t], false, false);
        }
    }

    // Epilogue: vn*tanh(x) = vn - 2*vn/(exp2(C*x)+1), C = 2*log2(e).
    // Sum of vn over the lane's 8 columns is hoisted (added once at the end).
    const float C = 2.885390082f;
    const int rbase = msub * 16 + ((lane < 16) ? 0 : 8);
    float epart[8];
#pragma unroll
    for (int j = 0; j < 8; ++j) epart[j] = 0.0f;
    float vsum = 0.0f;
#pragma unroll
    for (int t = 0; t < 8; ++t) {
        const int n    = (nhalf * 8 + t) * 16 + (lane & 15);
        const float hc   = hb[b * HH + n] * C;
        const float vn   = v[n];
        const float m2vn = -2.0f * vn;
        vsum += vn;
#pragma unroll
        for (int j = 0; j < 8; ++j) {
            const float ex = fast_exp2(fmaf(acc[t][j], C, hc));
            epart[j] = fmaf(m2vn, fast_rcp(ex + 1.0f), epart[j]);
        }
    }
#pragma unroll
    for (int j = 0; j < 8; ++j)
        atomicAdd(&e_lds[rbase + j], epart[j] + vsum);
    __syncthreads();
    if (threadIdx.x < 128)
        e_out[b * SS + s0 + threadIdx.x] = e_lds[threadIdx.x];
}

// ---------------------------------------------------------------------------
// Kernel 4: in-place softmax over S per batch row
// ---------------------------------------------------------------------------
__global__ __launch_bounds__(256) void softmax_kernel(float* __restrict__ e) {
    float* row = e + (size_t)blockIdx.x * SS;
    __shared__ float red[256];
    const int tid = threadIdx.x;
    float m = -INFINITY;
    for (int i = tid; i < SS; i += 256) m = fmaxf(m, row[i]);
    red[tid] = m; __syncthreads();
    for (int o = 128; o > 0; o >>= 1) {
        if (tid < o) red[tid] = fmaxf(red[tid], red[tid + o]);
        __syncthreads();
    }
    m = red[0]; __syncthreads();
    float sum = 0.0f;
    for (int i = tid; i < SS; i += 256) {
        float w = __expf(row[i] - m);
        row[i] = w;
        sum += w;
    }
    red[tid] = sum; __syncthreads();
    for (int o = 128; o > 0; o >>= 1) {
        if (tid < o) red[tid] += red[tid + o];
        __syncthreads();
    }
    const float inv = 1.0f / red[0];
    for (int i = tid; i < SS; i += 256) row[i] *= inv;
}

// ---------------------------------------------------------------------------
// Kernel 5a: zero output (we accumulate into it atomically)
// ---------------------------------------------------------------------------
__global__ __launch_bounds__(256) void zero_out_kernel(float* __restrict__ out) {
    out[blockIdx.x * 256 + threadIdx.x] = 0.0f;
}

// ---------------------------------------------------------------------------
// Kernel 5b: context[b,h] = sum_s w[b,s]*enc[b,s,h]; S chunked, f32 atomics.
// enc is L2-resident on the second pass (134 MB < 192 MB L2).
// ---------------------------------------------------------------------------
__global__ __launch_bounds__(256) void context_kernel(
    const float* __restrict__ enc, const float* __restrict__ w,
    float* __restrict__ out) {
    const int chunk = blockIdx.x;      // 0..31 (128 s each)
    const int b     = blockIdx.y;
    const int h     = threadIdx.x;
    const int sbeg  = chunk * 128;
    const float* erow = enc + ((size_t)b * SS + sbeg) * HH + h;
    const float* wrow = w + (size_t)b * SS + sbeg;
    float acc = 0.0f;
#pragma unroll 4
    for (int s = 0; s < 128; ++s)
        acc += wrow[s] * erow[(size_t)s * HH];
    atomicAdd(&out[b * HH + h], acc);
}

extern "C" void kernel_launch(void* const* d_in, const int* in_sizes, int n_in,
                              void* d_out, int out_size, void* d_ws, size_t ws_size,
                              hipStream_t stream) {
    const float* hidden = (const float*)d_in[0];   // [32,1,256]
    const float* enc    = (const float*)d_in[1];   // [32,4096,256]
    const float* W      = (const float*)d_in[2];   // [256,512]
    const float* b_attn = (const float*)d_in[3];   // [256]
    const float* v      = (const float*)d_in[4];   // [256]
    float* out = (float*)d_out;                    // [32,1,256]

    char* ws = (char*)d_ws;
    unsigned short* Wb = (unsigned short*)ws;                 // 128 KB
    float* hb = (float*)(ws + 128 * 1024);                    // 32 KB
    float* e  = (float*)(ws + 160 * 1024);                    // 512 KB

    repack_we_kernel<<<256, 256, 0, stream>>>(W, Wb);
    hidden_proj_kernel<<<BB, HH, 0, stream>>>(hidden, W, b_attn, hb);
    attn_energy_kernel<<<(BB * SS) / 128, 512, 0, stream>>>(enc, Wb, hb, v, e);
    softmax_kernel<<<BB, 256, 0, stream>>>(e);
    zero_out_kernel<<<(BB * HH) / 256, 256, 0, stream>>>(out);
    context_kernel<<<dim3(32, BB), HH, 0, stream>>>(enc, e, out);
}
